// EGAT_Rxn_40261023433289
// MI455X (gfx1250) — matrix operand order repaired and checked by reading, not compiled, not run on hardware
//
#include <hip/hip_runtime.h>

// ---------------------------------------------------------------------------
// CDNA5 (gfx1250) EGAT pipeline.
// Dense GEMMs -> v_wmma_f32_16x16x32_bf16 (bf16 in, f32 accumulate).
// Irregular graph ops (edge softmax, scatter-aggregate, segment readout)
// -> wave32 VALU kernels with f32 atomics (targets land in 192MB L2).
// ---------------------------------------------------------------------------

typedef __attribute__((ext_vector_type(16))) __bf16 v16bf;
typedef __attribute__((ext_vector_type(8)))  float  v8f;

union FragU { unsigned int u[8]; v16bf v; };

__device__ __forceinline__ unsigned short f2bf(float f) {
  unsigned int u = __float_as_uint(f);
  unsigned int r = u + 0x7fffu + ((u >> 16) & 1u);  // round-to-nearest-even
  return (unsigned short)(r >> 16);
}

__device__ __forceinline__ float gelu_f(float x) {
  return 0.5f * x * (1.0f + erff(x * 0.7071067811865475f));
}

__device__ __forceinline__ void atomicMaxF(float* addr, float val) {
  unsigned int* ua = (unsigned int*)addr;
  unsigned int old = *ua;
  while (__uint_as_float(old) < val) {
    unsigned int assumed = old;
    old = atomicCAS(ua, assumed, __float_as_uint(val));
    if (old == assumed) break;
  }
}

// ---------------------------------------------------------------------------
// Generic WMMA GEMM:  C[M,Nc] = act( A[M,K](bf16) * Bt[Nc,K](bf16)^T + bias )
// A row-major [M,K]; Bt is the weight matrix pre-transposed, row-major [Nc,K].
// Requirements: M%16==0, K%32==0, Nc%64==0. One wave computes a 16x64 tile.
// act: 0 = none, 1 = exact GELU.
// ---------------------------------------------------------------------------
__global__ __launch_bounds__(128)
void gemm_bf16_wmma(const unsigned short* __restrict__ A,
                    const unsigned short* __restrict__ Bt,
                    const float* __restrict__ bias,
                    float* __restrict__ C,
                    int M, int K, int Nc, int act) {
  const int lane  = threadIdx.x & 31;
  const int wave  = blockIdx.x * (blockDim.x >> 5) + (threadIdx.x >> 5);
  const int tilesN = Nc >> 6;
  const int tilesM = M >> 4;
  if (wave >= tilesM * tilesN) return;   // wave-uniform: EXEC stays all-ones
  const int tm = wave / tilesN;
  const int tn = wave % tilesN;
  const int row0 = tm << 4;
  const int col0 = tn << 6;
  const int half = lane >> 4;            // 0: lanes 0-15, 1: lanes 16-31
  const int lm   = lane & 15;

  v8f acc[4];
  v8f zero = {};
#pragma unroll
  for (int t = 0; t < 4; ++t) acc[t] = zero;

  const unsigned short* arow = A + (size_t)(row0 + lm) * K;

  for (int kk = 0; kk < K; kk += 32) {
    // ---- A fragment: 16x32 bf16, ISA layout:
    //      K = 16*(v>>2) + 8*half + 2*(v&3) (+0/+1 packed)
    FragU af;
#pragma unroll
    for (int v = 0; v < 8; ++v) {
      int k = kk + ((v >> 2) << 4) + (half << 3) + ((v & 3) << 1);
      af.u[v] = *(const unsigned int*)(arow + k);
    }
#pragma unroll
    for (int t = 0; t < 4; ++t) {
      // ---- B fragment: 32x16 bf16, ISA layout:
      //      lane n = col, K = 16*(v>>2) + 8*half + 2*(v&3) (+0/+1 packed)
      const unsigned short* brow = Bt + (size_t)(col0 + (t << 4) + lm) * K;
      FragU bf;
#pragma unroll
      for (int v = 0; v < 8; ++v) {
        int k = kk + ((v >> 2) << 4) + (half << 3) + ((v & 3) << 1);
        bf.u[v] = *(const unsigned int*)(brow + k);
      }
      acc[t] = __builtin_amdgcn_wmma_f32_16x16x32_bf16(
          false, af.v, false, bf.v, (short)0, acc[t], false, false);
    }
  }

  // ---- Epilogue: C/D layout -> row = row0 + r + 8*half, col = col0+t*16+lm
#pragma unroll
  for (int t = 0; t < 4; ++t) {
    int col = col0 + (t << 4) + lm;
    float b = bias ? bias[col] : 0.0f;
#pragma unroll
    for (int r = 0; r < 8; ++r) {
      int row = row0 + r + (half << 3);
      float x = acc[t][r] + b;
      if (act == 1) x = gelu_f(x);
      C[(size_t)row * Nc + col] = x;
    }
  }
}

// ---------------------------------------------------------------------------
// f32 -> bf16 with K padding: out[r*kpad+k] = k<kin ? bf16(in[r*kin+k]) : 0
// ---------------------------------------------------------------------------
__global__ __launch_bounds__(256)
void f32_to_bf16_pad(const float* __restrict__ in, unsigned short* __restrict__ out,
                     int rows, int kin, int kpad) {
  size_t i = (size_t)blockIdx.x * blockDim.x + threadIdx.x;
  size_t n = (size_t)rows * kpad;
  if (i >= n) return;
  int r = (int)(i / kpad), k = (int)(i % kpad);
  out[i] = (k < kin) ? f2bf(in[(size_t)r * kin + k]) : (unsigned short)0;
}

// Weight convert + transpose + pad: W[kin,ncols] f32 -> Wt[ncols,kpad] bf16
__global__ __launch_bounds__(256)
void wt_convert(const float* __restrict__ W, unsigned short* __restrict__ Wt,
                int kin, int kpad, int ncols) {
  size_t i = (size_t)blockIdx.x * blockDim.x + threadIdx.x;
  size_t n = (size_t)ncols * kpad;
  if (i >= n) return;
  int nn = (int)(i / kpad), k = (int)(i % kpad);
  Wt[i] = (k < kin) ? f2bf(W[(size_t)k * ncols + nn]) : (unsigned short)0;
}

__global__ __launch_bounds__(256)
void fill_kernel(float* __restrict__ p, float v, size_t n) {
  size_t i = (size_t)blockIdx.x * blockDim.x + threadIdx.x;
  if (i < n) p[i] = v;
}

// bias_cat[768]: 0 for cols 0..511 (PNi|PNj), b_node for cols 512..767 (PH)
__global__ void biascat_kernel(const float* __restrict__ bnode, float* __restrict__ out) {
  int i = blockIdx.x * 256 + threadIdx.x;
  if (i < 768) out[i] = (i < 512) ? 0.0f : bnode[i - 512];
}

// ---------------------------------------------------------------------------
// Edge logits: one wave per edge. f = leaky(PNi[src]+PNj[dst]+PE[e]);
// store f (next ef), per-head dot with attn, atomic segment max over dst.
// ---------------------------------------------------------------------------
__global__ __launch_bounds__(256)
void edge_logits_kernel(const float* __restrict__ PN, const float* __restrict__ PE,
                        const int* __restrict__ src, const int* __restrict__ dst,
                        const float* __restrict__ attn,
                        float* __restrict__ fout, float* __restrict__ logit,
                        float* __restrict__ mbuf, int E_) {
  int e = blockIdx.x * 8 + (threadIdx.x >> 5);
  if (e >= E_) return;
  int lane = threadIdx.x & 31;
  int s = src[e], d = dst[e];
  const float* pni = PN + (size_t)s * 768;
  const float* pnj = PN + (size_t)d * 768 + 256;
  const float* pe  = PE + (size_t)e * 256;
  float* fo        = fout + (size_t)e * 256;
  int c0 = lane << 3;                 // 8 consecutive cols/lane, within one head
  float part = 0.0f;
#pragma unroll
  for (int j = 0; j < 8; ++j) {
    int c = c0 + j;
    float x = pni[c] + pnj[c] + pe[c];
    x = (x > 0.0f) ? x : 0.01f * x;   // leaky_relu(0.01)
    fo[c] = x;
    part += x * attn[c];
  }
  part += __shfl_xor(part, 1, 32);
  part += __shfl_xor(part, 2, 32);
  part += __shfl_xor(part, 4, 32);    // reduce 8-lane head group
  if ((lane & 7) == 0) {
    int h = lane >> 3;
    logit[(size_t)e * 4 + h] = part;
    atomicMaxF(&mbuf[(size_t)d * 4 + h], part);
  }
}

// w = exp(e - m[dst]); in-place overwrite; atomic segment sum
__global__ __launch_bounds__(256)
void softmax_w_kernel(float* __restrict__ logit, const float* __restrict__ mbuf,
                      float* __restrict__ sbuf, const int* __restrict__ dst, int n) {
  int i = blockIdx.x * 256 + threadIdx.x;
  if (i >= n) return;
  int e = i >> 2, h = i & 3;
  int d = dst[e];
  float w = expf(logit[i] - mbuf[(size_t)d * 4 + h]);
  logit[i] = w;
  atomicAdd(&sbuf[(size_t)d * 4 + h], w);
}

// nf_out[dst] += PH[src] * a  (a = w / s[dst]); one wave per edge
__global__ __launch_bounds__(256)
void aggregate_kernel(const float* __restrict__ PN, const float* __restrict__ logit,
                      const float* __restrict__ sbuf,
                      const int* __restrict__ src, const int* __restrict__ dst,
                      float* __restrict__ nfout, int E_) {
  int e = blockIdx.x * 8 + (threadIdx.x >> 5);
  if (e >= E_) return;
  int lane = threadIdx.x & 31;
  int s = src[e], d = dst[e];
  int h = lane >> 3;
  float a = logit[(size_t)e * 4 + h] / sbuf[(size_t)d * 4 + h];
  const float* ph = PN + (size_t)s * 768 + 512;
  float* no = nfout + (size_t)d * 256;
  int c0 = lane << 3;
#pragma unroll
  for (int j = 0; j < 8; ++j) {
    int c = c0 + j;
    atomicAdd(&no[c], ph[c] * a);
  }
}

// out[seg[r]*256+c] += x[r*256+c]
__global__ __launch_bounds__(256)
void segsum_kernel(const float* __restrict__ x, const int* __restrict__ seg,
                   float* __restrict__ out, int rows) {
  size_t i = (size_t)blockIdx.x * blockDim.x + threadIdx.x;
  if (i >= (size_t)rows * 256) return;
  int r = (int)(i >> 8), c = (int)(i & 255);
  atomicAdd(&out[(size_t)seg[r] * 256 + c], x[i]);
}

// G = bf16(concat(Gn, Ge)) -> [B,512]
__global__ __launch_bounds__(256)
void concat_bf16_kernel(const float* __restrict__ Gn, const float* __restrict__ Ge,
                        unsigned short* __restrict__ G, int Brows) {
  int i = blockIdx.x * 256 + threadIdx.x;
  if (i >= Brows * 512) return;
  int r = i >> 9, c = i & 511;
  float v = (c < 256) ? Gn[(size_t)r * 256 + c] : Ge[(size_t)r * 256 + (c - 256)];
  G[i] = f2bf(v);
}

// out[b] = dot(x2[b,:128], W3) + b3
__global__ __launch_bounds__(128)
void mlp3_kernel(const float* __restrict__ x2, const float* __restrict__ W3,
                 const float* __restrict__ b3, float* __restrict__ out, int Brows) {
  __shared__ float red[128];
  int b = blockIdx.x, t = threadIdx.x;
  red[t] = x2[(size_t)b * 128 + t] * W3[t];
  __syncthreads();
  for (int s = 64; s > 0; s >>= 1) {
    if (t < s) red[t] += red[t + s];
    __syncthreads();
  }
  if (t == 0) out[b] = red[0] + b3[0];
}

// ---------------------------------------------------------------------------
extern "C" void kernel_launch(void* const* d_in, const int* in_sizes, int n_in,
                              void* d_out, int out_size, void* d_ws, size_t ws_size,
                              hipStream_t stream) {
  const int N = 100000, E = 400000, Bg = 4096, HD = 256;

  const float* node_feats = (const float*)d_in[0];
  const float* edge_feats = (const float*)d_in[1];
  const int*   src        = (const int*)d_in[3];
  const int*   dst        = (const int*)d_in[4];
  const int*   node_seg   = (const int*)d_in[5];
  const int*   edge_seg   = (const int*)d_in[6];
  const float* W1_ni   = (const float*)d_in[7];
  const float* W1_nj   = (const float*)d_in[8];
  const float* W1_fij  = (const float*)d_in[9];
  const float* attn1   = (const float*)d_in[10];
  const float* W1_node = (const float*)d_in[11];
  const float* b1_node = (const float*)d_in[12];
  const float* W2_ni   = (const float*)d_in[13];
  const float* W2_nj   = (const float*)d_in[14];
  const float* W2_fij  = (const float*)d_in[15];
  const float* attn2   = (const float*)d_in[16];
  const float* W2_node = (const float*)d_in[17];
  const float* b2_node = (const float*)d_in[18];
  const float* aggN_W  = (const float*)d_in[19];
  const float* aggN_b  = (const float*)d_in[20];
  const float* aggE_W  = (const float*)d_in[21];
  const float* aggE_b  = (const float*)d_in[22];
  const float* mlp1_W  = (const float*)d_in[23];
  const float* mlp1_b  = (const float*)d_in[24];
  const float* mlp2_W  = (const float*)d_in[25];
  const float* mlp2_b  = (const float*)d_in[26];
  const float* mlp3_W  = (const float*)d_in[27];
  const float* mlp3_b  = (const float*)d_in[28];

  char* ws = (char*)d_ws;
  size_t off = 0;
  auto alloc = [&](size_t bytes) -> char* {
    char* p = ws + off;
    off += (bytes + 255) & ~(size_t)255;
    return p;
  };

  float*          nf    = (float*)alloc((size_t)N * HD * 4);   // node feats / aggregate out
  float*          efA   = (float*)alloc((size_t)E * HD * 4);   // f_out (edge feats)
  float*          efB   = (float*)alloc((size_t)E * HD * 4);   // PE / agg-edge out
  float*          PN    = (float*)alloc((size_t)N * 768 * 4);  // PNi|PNj|PH
  unsigned short* abf   = (unsigned short*)alloc((size_t)E * HD * 2);  // shared bf16 acts
  float*          logit = (float*)alloc((size_t)E * 4 * 4);
  float*          mbuf  = (float*)alloc((size_t)N * 4 * 4);
  float*          sbuf  = (float*)alloc((size_t)N * 4 * 4);
  float*          Gn    = (float*)alloc((size_t)Bg * 256 * 4);
  float*          Ge    = (float*)alloc((size_t)Bg * 256 * 4);
  unsigned short* Gbf   = (unsigned short*)alloc((size_t)Bg * 512 * 2);
  float*          x1    = (float*)alloc((size_t)Bg * 256 * 4);
  unsigned short* x1bf  = (unsigned short*)alloc((size_t)Bg * 256 * 2);
  float*          x2    = (float*)alloc((size_t)Bg * 128 * 4);
  float*          biascat = (float*)alloc(768 * 4);
  unsigned short* Wcat1 = (unsigned short*)alloc((size_t)768 * 32 * 2);
  unsigned short* Wfij1 = (unsigned short*)alloc((size_t)256 * 32 * 2);
  unsigned short* Wcat2 = (unsigned short*)alloc((size_t)768 * 256 * 2);
  unsigned short* Wfij2 = (unsigned short*)alloc((size_t)256 * 256 * 2);
  unsigned short* WaggN = (unsigned short*)alloc((size_t)256 * 256 * 2);
  unsigned short* WaggE = (unsigned short*)alloc((size_t)256 * 256 * 2);
  unsigned short* Wm1   = (unsigned short*)alloc((size_t)256 * 512 * 2);
  unsigned short* Wm2   = (unsigned short*)alloc((size_t)128 * 256 * 2);
  if (off > ws_size) return;  // workspace too small: bail rather than corrupt

  auto wtconv = [&](const float* W, unsigned short* Wt, int kin, int kpad, int ncols) {
    size_t n = (size_t)ncols * kpad;
    wt_convert<<<(unsigned)((n + 255) / 256), 256, 0, stream>>>(W, Wt, kin, kpad, ncols);
  };
  auto fill = [&](float* p, float v, size_t n) {
    fill_kernel<<<(unsigned)((n + 255) / 256), 256, 0, stream>>>(p, v, n);
  };
  auto conv = [&](const float* in, unsigned short* out, int rows, int kin, int kpad) {
    size_t n = (size_t)rows * kpad;
    f32_to_bf16_pad<<<(unsigned)((n + 255) / 256), 256, 0, stream>>>(in, out, rows, kin, kpad);
  };
  auto gemm = [&](const unsigned short* A, const unsigned short* Bt, const float* bias,
                  float* C, int M, int K, int Nc, int act) {
    int waves = (M >> 4) * (Nc >> 6);
    gemm_bf16_wmma<<<(unsigned)((waves + 3) / 4), 128, 0, stream>>>(A, Bt, bias, C, M, K, Nc, act);
  };

  // ---- weight prep (bf16, transposed, K padded) ----
  wtconv(W1_ni,   Wcat1,             17, 32, 256);
  wtconv(W1_nj,   Wcat1 + 256 * 32,  17, 32, 256);
  wtconv(W1_node, Wcat1 + 512 * 32,  17, 32, 256);
  wtconv(W1_fij,  Wfij1,             15, 32, 256);
  wtconv(W2_ni,   Wcat2,             256, 256, 256);
  wtconv(W2_nj,   Wcat2 + 256 * 256, 256, 256, 256);
  wtconv(W2_node, Wcat2 + 512 * 256, 256, 256, 256);
  wtconv(W2_fij,  Wfij2,             256, 256, 256);
  wtconv(aggN_W,  WaggN,             256, 256, 256);
  wtconv(aggE_W,  WaggE,             256, 256, 256);
  wtconv(mlp1_W,  Wm1,               512, 512, 256);
  wtconv(mlp2_W,  Wm2,               256, 256, 128);

  auto layer = [&](const float* nf_in, int nKin, int nKpad,
                   const float* ef_in, int eKin, int eKpad,
                   const unsigned short* WcatT, const unsigned short* WfijT,
                   const float* attn) {
    conv(nf_in, abf, N, nKin, nKpad);
    gemm(abf, WcatT, biascat, PN, N, nKpad, 768, 0);     // PNi|PNj|PH(+b_node)
    conv(ef_in, abf, E, eKin, eKpad);
    gemm(abf, WfijT, nullptr, efB, E, eKpad, 256, 0);    // PE
    fill(mbuf, -3.0e38f, (size_t)N * 4);
    fill(sbuf, 0.0f, (size_t)N * 4);
    edge_logits_kernel<<<E / 8, 256, 0, stream>>>(PN, efB, src, dst, attn,
                                                  efA, logit, mbuf, E);
    softmax_w_kernel<<<(E * 4 + 255) / 256, 256, 0, stream>>>(logit, mbuf, sbuf, dst, E * 4);
    fill(nf, 0.0f, (size_t)N * HD);
    aggregate_kernel<<<E / 8, 256, 0, stream>>>(PN, logit, sbuf, src, dst, nf, E);
  };

  // ---- 3 EGAT layers (layer 1 with W1, layers 2-3 share W2) ----
  biascat_kernel<<<3, 256, 0, stream>>>(b1_node, biascat);
  layer(node_feats, 17, 32, edge_feats, 15, 32, Wcat1, Wfij1, attn1);
  biascat_kernel<<<3, 256, 0, stream>>>(b2_node, biascat);
  layer(nf, 256, 256, efA, 256, 256, Wcat2, Wfij2, attn2);
  layer(nf, 256, 256, efA, 256, 256, Wcat2, Wfij2, attn2);

  // ---- readout: gelu(agg) then per-graph segment sums ----
  conv(nf, abf, N, 256, 256);
  gemm(abf, WaggN, aggN_b, PN, N, 256, 256, 1);          // reuse PN as [N,256]
  fill(Gn, 0.0f, (size_t)Bg * 256);
  segsum_kernel<<<(unsigned)(((size_t)N * 256 + 255) / 256), 256, 0, stream>>>(PN, node_seg, Gn, N);

  conv(efA, abf, E, 256, 256);
  gemm(abf, WaggE, aggE_b, efB, E, 256, 256, 1);
  fill(Ge, 0.0f, (size_t)Bg * 256);
  segsum_kernel<<<(unsigned)(((size_t)E * 256 + 255) / 256), 256, 0, stream>>>(efB, edge_seg, Ge, E);

  // ---- MLP head ----
  concat_bf16_kernel<<<(Bg * 512 + 255) / 256, 256, 0, stream>>>(Gn, Ge, Gbf, Bg);
  gemm(Gbf, Wm1, mlp1_b, x1, Bg, 512, 256, 1);
  conv(x1, x1bf, Bg, 256, 256);
  gemm(x1bf, Wm2, mlp2_b, x2, Bg, 256, 128, 1);
  mlp3_kernel<<<Bg, 128, 0, stream>>>(x2, mlp3_W, mlp3_b, (float*)d_out, Bg);

  (void)in_sizes; (void)n_in; (void)out_size;
}
